// ProteinPocketGNN_28260884807710
// MI455X (gfx1250) — matrix operand order repaired and checked by reading, not compiled, not run on hardware
//
#include <hip/hip_runtime.h>
#include <hip/hip_bf16.h>

typedef __attribute__((ext_vector_type(16))) _Float16 v16h;
typedef __attribute__((ext_vector_type(8)))  _Float16 v8h;
typedef __attribute__((ext_vector_type(8)))  float    v8f;

#define BN_EPS 1e-5f

// ---------------------------------------------------------------- utilities
__global__ void zero_f32(float* p, long long n) {
  long long i = (long long)blockIdx.x * blockDim.x + threadIdx.x;
  if (i < n) p[i] = 0.0f;
}

// ---------------------------------------------------------------- degree / norm
__global__ void deg_accum(const int* __restrict__ dst, float* __restrict__ deg, int E) {
  int e = blockIdx.x * blockDim.x + threadIdx.x;
  if (e < E) atomicAdd(&deg[dst[e]], 1.0f);
}

// in-place: dis[] holds raw degree on entry; writes dis = (deg+1)^-1/2, dinv = 1/(deg+1)
__global__ void deg_norm(float* __restrict__ dis, float* __restrict__ dinv, int N) {
  int i = blockIdx.x * blockDim.x + threadIdx.x;
  if (i < N) {
    float d = dis[i] + 1.0f;
    dis[i]  = rsqrtf(d);
    dinv[i] = 1.0f / d;
  }
}

__global__ void edge_norm(const int* __restrict__ src, const int* __restrict__ dst,
                          const float* __restrict__ dis, float* __restrict__ ne, int E) {
  int e = blockIdx.x * blockDim.x + threadIdx.x;
  if (e < E) ne[e] = dis[src[e]] * dis[dst[e]];
}

// ---------------------------------------------------------------- WMMA GEMM
// out[N,FOUT] = X[N,K] @ W[K,FOUT].  f16 operands, f32 accumulate.
// Block = 256 threads = 8 waves; block covers 64 rows x FOUT cols.
// X tile and transposed W staged in LDS as f16 (zero-padded K -> KP), so all
// fragment loads are unconditional 16B ds_load_b128s and WMMA runs with no
// per-lane guards (EXEC all-1s throughout).
template<int K, int FOUT>
__global__ __launch_bounds__(256) void gemm_wmma(const float* __restrict__ X,
                                                 const float* __restrict__ W,
                                                 float* __restrict__ out, int N) {
  constexpr int KP      = (K + 31) & ~31;   // 32 or 64
  constexpr int XST     = KP + 8;           // row stride in halfs (16B aligned, bank-spread)
  constexpr int WST     = KP + 8;
  constexpr int TILES_N = FOUT / 16;
  constexpr int TILES   = 4 * TILES_N;      // 4 row-subtiles x TILES_N col-subtiles

  __shared__ _Float16 Xs[64 * XST];
  __shared__ _Float16 Wt[FOUT * WST];       // Wt[n][k] = W[k][n]

  const int t    = threadIdx.x;
  const int base = blockIdx.x * 64;

  // ---- stage X tile (rows clamped: clamped rows feed only never-stored tiles)
  for (int idx = t; idx < 64 * KP; idx += 256) {
    int r = idx / KP, k = idx % KP;
    int rg = base + r;
    rg = rg < N ? rg : N - 1;
    float v = (k < K) ? X[(long long)rg * K + k] : 0.0f;
    Xs[r * XST + k] = (_Float16)v;
  }
  // ---- stage W transposed (pad columns zeroed)
  for (int idx = t; idx < FOUT * KP; idx += 256) {
    int n = idx / KP, k = idx % KP;
    float v = (k < K) ? W[k * FOUT + n] : 0.0f;
    Wt[n * WST + k] = (_Float16)v;
  }
  __syncthreads();

  const int wave = t >> 5;
  const int lane = t & 31;
  const int m    = lane & 15;               // A row / B,D column within tile
  const int hk   = lane >> 4;               // half-wave selector

  for (int tt = wave; tt < TILES; tt += 8) {
    const int sm = tt / TILES_N;
    const int sn = tt % TILES_N;
    const _Float16* xrow = &Xs[(sm * 16 + m) * XST];
    const _Float16* wrow = &Wt[(sn * 16 + m) * WST];

    v8f c;
    #pragma unroll
    for (int r = 0; r < 8; ++r) c[r] = 0.0f;

    #pragma unroll
    for (int kb = 0; kb < KP; kb += 32) {
      // A fragment (ISA 7.12.2): lane holds K = kb + hk*8 + [0,8) and kb + 16 + hk*8 + [0,8)
      v8h alo = *(const v8h*)(xrow + kb + hk * 8);
      v8h ahi = *(const v8h*)(xrow + kb + 16 + hk * 8);
      // B fragment: lane holds K = kb + hk*8 + [0,8) and kb + 16 + hk*8 + [0,8), column n = lane%16
      v8h blo = *(const v8h*)(wrow + kb + hk * 8);
      v8h bhi = *(const v8h*)(wrow + kb + 16 + hk * 8);
      v16h a, b;
      #pragma unroll
      for (int i = 0; i < 8; ++i) {
        a[i] = alo[i]; a[i + 8] = ahi[i];
        b[i] = blo[i]; b[i + 8] = bhi[i];
      }
      c = __builtin_amdgcn_wmma_f32_16x16x32_f16(false, a, false, b, (short)0, c,
                                                 false, false);
    }

    // D layout: VGPR r -> row (r + 8*hk), col = lane%16
    const int rowBase = base + sm * 16;
    if (rowBase + 16 <= N) {
      float* op = out + (long long)(rowBase + hk * 8) * FOUT + sn * 16 + m;
      #pragma unroll
      for (int r = 0; r < 8; ++r) op[(long long)r * FOUT] = c[r];
    } else {
      #pragma unroll
      for (int r = 0; r < 8; ++r) {
        int orow = rowBase + hk * 8 + r;
        if (orow < N) out[(long long)orow * FOUT + sn * 16 + m] = c[r];
      }
    }
  }
}

// ---------------------------------------------------------------- edge scatter
// agg[dst] += hW[src] * norm_e.  Thread = (edge, 4-feature group); consecutive
// threads cover consecutive groups of one edge -> coalesced gather + atomics.
template<int F>
__global__ void scatter_agg(const float* __restrict__ hW, const int* __restrict__ src,
                            const int* __restrict__ dst, const float* __restrict__ ne,
                            float* __restrict__ agg, int E) {
  const int G = F / 4;
  long long idx = (long long)blockIdx.x * blockDim.x + threadIdx.x;
  if (idx >= (long long)E * G) return;
  int e = (int)(idx / G);
  int g = (int)(idx % G) * 4;
  float nrm = ne[e];
  const float4 hv = *(const float4*)(hW + (long long)src[e] * F + g);
  float* p = agg + (long long)dst[e] * F + g;
  atomicAdd(p + 0, hv.x * nrm);
  atomicAdd(p + 1, hv.y * nrm);
  atomicAdd(p + 2, hv.z * nrm);
  atomicAdd(p + 3, hv.w * nrm);
}

// ---------------------------------------------------------------- epilogues
template<int F>
__global__ void epi_bn_relu(const float* __restrict__ agg, const float* __restrict__ hW,
                            const float* __restrict__ dinv, const float* __restrict__ bias,
                            const float* __restrict__ gam, const float* __restrict__ bet,
                            const float* __restrict__ mean, const float* __restrict__ var,
                            float* __restrict__ hout, int N) {
  long long idx = (long long)blockIdx.x * blockDim.x + threadIdx.x;
  if (idx >= (long long)N * F) return;
  int i = (int)(idx / F);
  int f = (int)(idx % F);
  float val = agg[idx] + hW[idx] * dinv[i] + bias[f];
  val = (val - mean[f]) * rsqrtf(var[f] + BN_EPS) * gam[f] + bet[f];
  hout[idx] = fmaxf(val, 0.0f);
}

template<int F>
__global__ void epi_relu(const float* __restrict__ agg, const float* __restrict__ hW,
                         const float* __restrict__ dinv, const float* __restrict__ bias,
                         float* __restrict__ hout, int N) {
  long long idx = (long long)blockIdx.x * blockDim.x + threadIdx.x;
  if (idx >= (long long)N * F) return;
  int i = (int)(idx / F);
  int f = (int)(idx % F);
  float val = agg[idx] + hW[idx] * dinv[i] + bias[f];
  hout[idx] = fmaxf(val, 0.0f);
}

// ---------------------------------------------------------------- pooling
__global__ void pool_accum(const float* __restrict__ h, const int* __restrict__ batch,
                           float* __restrict__ sums, float* __restrict__ cnt, int N) {
  const int G = 128 / 4;
  long long idx = (long long)blockIdx.x * blockDim.x + threadIdx.x;
  if (idx >= (long long)N * G) return;
  int i = (int)(idx / G);
  int g = (int)(idx % G) * 4;
  int b = batch[i];
  const float4 hv = *(const float4*)(h + (long long)i * 128 + g);
  float* p = sums + (long long)b * 128 + g;
  atomicAdd(p + 0, hv.x);
  atomicAdd(p + 1, hv.y);
  atomicAdd(p + 2, hv.z);
  atomicAdd(p + 3, hv.w);
  if (g == 0) atomicAdd(&cnt[b], 1.0f);
}

__global__ void pool_final(const float* __restrict__ sums, const float* __restrict__ cnt,
                           float* __restrict__ out, int B) {
  int idx = blockIdx.x * blockDim.x + threadIdx.x;
  if (idx < B * 128) {
    int b = idx >> 7;
    out[idx] = sums[idx] / fmaxf(cnt[b], 1.0f);
  }
}

// ---------------------------------------------------------------- launch
static inline int cdiv(long long a, long long b) { return (int)((a + b - 1) / b); }

extern "C" void kernel_launch(void* const* d_in, const int* in_sizes, int n_in,
                              void* d_out, int out_size, void* d_ws, size_t ws_size,
                              hipStream_t stream) {
  const float* x   = (const float*)d_in[0];
  const int*   ei  = (const int*)d_in[1];
  const int*   bat = (const int*)d_in[2];
  const float* W1 = (const float*)d_in[3],  *b1 = (const float*)d_in[4];
  const float* W2 = (const float*)d_in[5],  *b2 = (const float*)d_in[6];
  const float* W3 = (const float*)d_in[7],  *b3 = (const float*)d_in[8];
  const float* g1 = (const float*)d_in[9],  *be1 = (const float*)d_in[10];
  const float* m1 = (const float*)d_in[11], *v1  = (const float*)d_in[12];
  const float* g2 = (const float*)d_in[13], *be2 = (const float*)d_in[14];
  const float* m2 = (const float*)d_in[15], *v2  = (const float*)d_in[16];

  const int N = in_sizes[2];
  const int E = in_sizes[1] / 2;
  const int B = out_size / 128;
  const int* src = ei;
  const int* dst = ei + E;

  // workspace carve-up (64-float aligned chunks)
  auto align64 = [](size_t v) { return (v + 63) & ~(size_t)63; };
  float* wsf  = (float*)d_ws;
  size_t off = 0;
  float* dis  = wsf + off; off += align64(N);
  float* dinv = wsf + off; off += align64(N);
  float* ne   = wsf + off; off += align64(E);
  float* hW   = wsf + off; off += align64((size_t)N * 128);
  float* agg  = wsf + off; off += align64((size_t)N * 128);
  float* h    = wsf + off; off += align64((size_t)N * 128);
  float* sums = wsf + off; off += align64((size_t)B * 128);
  float* cnt  = wsf + off; off += align64(B);
  (void)ws_size; (void)n_in;

  const int T = 256;                // 8 waves per block

  // ---- degree + symmetric norm
  zero_f32<<<cdiv(N, T), T, 0, stream>>>(dis, N);
  deg_accum<<<cdiv(E, T), T, 0, stream>>>(dst, dis, E);
  deg_norm<<<cdiv(N, T), T, 0, stream>>>(dis, dinv, N);
  edge_norm<<<cdiv(E, T), T, 0, stream>>>(src, dst, dis, ne, E);

  // ---- layer 1: 22 -> 64, BN, ReLU
  {
    gemm_wmma<22, 64><<<cdiv(N, 64), T, 0, stream>>>(x, W1, hW, N);
    zero_f32<<<cdiv((long long)N * 64, T), T, 0, stream>>>(agg, (long long)N * 64);
    scatter_agg<64><<<cdiv((long long)E * 16, T), T, 0, stream>>>(hW, src, dst, ne, agg, E);
    epi_bn_relu<64><<<cdiv((long long)N * 64, T), T, 0, stream>>>(
        agg, hW, dinv, b1, g1, be1, m1, v1, h, N);
  }
  // ---- layer 2: 64 -> 64, BN, ReLU  (h is both input and output; stream-ordered)
  {
    gemm_wmma<64, 64><<<cdiv(N, 64), T, 0, stream>>>(h, W2, hW, N);
    zero_f32<<<cdiv((long long)N * 64, T), T, 0, stream>>>(agg, (long long)N * 64);
    scatter_agg<64><<<cdiv((long long)E * 16, T), T, 0, stream>>>(hW, src, dst, ne, agg, E);
    epi_bn_relu<64><<<cdiv((long long)N * 64, T), T, 0, stream>>>(
        agg, hW, dinv, b2, g2, be2, m2, v2, h, N);
  }
  // ---- layer 3: 64 -> 128, ReLU
  {
    gemm_wmma<64, 128><<<cdiv(N, 64), T, 0, stream>>>(h, W3, hW, N);
    zero_f32<<<cdiv((long long)N * 128, T), T, 0, stream>>>(agg, (long long)N * 128);
    scatter_agg<128><<<cdiv((long long)E * 32, T), T, 0, stream>>>(hW, src, dst, ne, agg, E);
    epi_relu<128><<<cdiv((long long)N * 128, T), T, 0, stream>>>(agg, hW, dinv, b3, h, N);
  }
  // ---- global mean pool
  zero_f32<<<cdiv((long long)B * 128, T), T, 0, stream>>>(sums, (long long)B * 128);
  zero_f32<<<cdiv(B, T), T, 0, stream>>>(cnt, B);
  pool_accum<<<cdiv((long long)N * 32, T), T, 0, stream>>>(h, bat, sums, cnt, N);
  pool_final<<<cdiv((long long)B * 128, T), T, 0, stream>>>(sums, cnt, (float*)d_out, B);
}
